// GetContext_5136780886780
// MI455X (gfx1250) — matrix operand order repaired and checked by reading, not compiled, not run on hardware
//
#include <hip/hip_runtime.h>
#include <hip/hip_bf16.h>
#include <math.h>

// ---------------- problem constants ----------------
#define N_NODES 20000
#define N_EDGES 320000
#define F_NODE  128
#define F_EDGE  16
#define GDIM    256
#define KE1     160   // 144 padded up to multiple of 32
#define KE1_TRUE 144

typedef __attribute__((ext_vector_type(16))) _Float16 v16h;
typedef __attribute__((ext_vector_type(8)))  _Float16 v8h;
typedef __attribute__((ext_vector_type(8)))  float    v8f;

// ---------------- device helpers ----------------
static __device__ __forceinline__ float lrelu(float x) { return x >= 0.f ? x : 0.01f * x; }
static __device__ __forceinline__ float sigmoidf(float x) { return 1.f / (1.f + __expf(-x)); }

// order-preserving float -> uint encoding for atomic max
static __device__ __forceinline__ unsigned fenc(float f) {
    unsigned u = __float_as_uint(f);
    return (u & 0x80000000u) ? ~u : (u | 0x80000000u);
}
static __device__ __forceinline__ float fdec(unsigned k) {
    if (k == 0u) return 0.f;   // node with no incoming edges -> reference maps -inf to 0
    unsigned u = (k & 0x80000000u) ? (k & 0x7FFFFFFFu) : ~k;
    return __uint_as_float(u);
}

// D = A*B + C, 16x16x32 f16 -> f32 accumulate (codegen-confirmed builtin)
static __device__ __forceinline__ v8f wmma_f16(v16h a, v16h b, v8f c) {
    return __builtin_amdgcn_wmma_f32_16x16x32_f16(false, a, false, b, (short)0, c, false, false);
}

// Assemble A-fragment (16x32 f16, row m = lane&15) from row-major f16 buffer.
// Element e of v16h holds K = kbase + e + (e>=8 ? 8 : 0) + (lane>=16 ? 8 : 0):
// two contiguous 8-half (16B) runs.
static __device__ __forceinline__ v16h load_a_frag(const _Float16* rowk, int lane) {
    int off = (lane & 16) ? 8 : 0;
    v8h lo = *(const v8h*)(rowk + off);
    v8h hi = *(const v8h*)(rowk + 16 + off);
    return __builtin_shufflevector(lo, hi, 0,1,2,3,4,5,6,7,8,9,10,11,12,13,14,15);
}

// ---------------- weight packing ----------------
// Fragment-major B layout: frag (kt, nt) of a K x N matrix stored as 32 lanes x 16 f16
// contiguous; lane holds N = nt*16 + (lane&15), element e holds K = kt*32 + (lane>=16?8:0) + (e&7) + (e>=8?16:0).
__global__ void k_pack_b(const float* __restrict__ W, _Float16* __restrict__ out,
                         int Kpad, int Ktrue, int N, int transpose) {
    int t = blockIdx.x * blockDim.x + threadIdx.x;
    int total = Kpad * N;
    if (t >= total) return;
    int ntiles = N >> 4;
    int fragId = t >> 9;
    int within = t & 511;
    int lane = within >> 4;
    int e = within & 15;
    int kt = fragId / ntiles;
    int nt = fragId - kt * ntiles;
    int k = kt * 32 + ((lane & 16) ? 8 : 0) + (e & 7) + ((e & 8) ? 16 : 0);
    int n = nt * 16 + (lane & 15);
    float v = 0.f;
    if (k < Ktrue) v = transpose ? W[(size_t)n * Ktrue + k] : W[(size_t)k * N + n];
    out[t] = (_Float16)v;
}

__global__ void k_cvt_f16(const float* __restrict__ in, _Float16* __restrict__ out, int n) {
    int i = blockIdx.x * blockDim.x + threadIdx.x;
    if (i < n) out[i] = (_Float16)in[i];
}

__global__ void k_init(unsigned* __restrict__ segmaxK, float* __restrict__ segsum,
                       float* __restrict__ c) {
    int i = blockIdx.x * blockDim.x + threadIdx.x;
    if (i < N_NODES * GDIM) c[i] = 0.f;
    if (i < N_NODES) { segmaxK[i] = 0u; segsum[i] = 0.f; }
}

// ---------------- stage 1: hv_new = leaky_relu(node @ W_node + b) ----------------
__global__ void __launch_bounds__(256) k_node_gemm(const _Float16* __restrict__ nodeF16,
                                                   const _Float16* __restrict__ WnodeP,
                                                   const float* __restrict__ b_node,
                                                   float* __restrict__ hv) {
    __shared__ _Float16 As[16 * F_NODE];
    int tile = blockIdx.x, t = threadIdx.x;
    for (int i = t; i < 16 * F_NODE; i += 256) {
        int r = i >> 7, col = i & 127;
        As[i] = nodeF16[(size_t)(tile * 16 + r) * F_NODE + col];
    }
    __syncthreads();
    int wave = t >> 5, lane = t & 31;
    const _Float16* arow = &As[(lane & 15) * F_NODE];
    v16h afrag[4];
#pragma unroll
    for (int kt = 0; kt < 4; ++kt) afrag[kt] = load_a_frag(arow + kt * 32, lane);
    const v16h* Bp = (const v16h*)WnodeP;
#pragma unroll
    for (int ni = 0; ni < 2; ++ni) {
        int nt = wave * 2 + ni;
        v8f acc = {0.f,0.f,0.f,0.f,0.f,0.f,0.f,0.f};
#pragma unroll
        for (int kt = 0; kt < 4; ++kt)
            acc = wmma_f16(afrag[kt], Bp[(kt * 16 + nt) * 32 + lane], acc);
        int n = nt * 16 + (lane & 15);
        float bias = b_node[n];
#pragma unroll
        for (int r = 0; r < 8; ++r) {
            int m = tile * 16 + r + ((lane & 16) ? 8 : 0);
            hv[(size_t)m * GDIM + n] = lrelu(acc[r] + bias);
        }
    }
}

// ---------------- stage 2+3: he1, attention logits, segment max ----------------
__global__ void __launch_bounds__(256) k_edge_he1(const _Float16* __restrict__ nodeF16,
                                                  const float* __restrict__ edge_feats,
                                                  const int* __restrict__ src,
                                                  const int* __restrict__ dst,
                                                  const _Float16* __restrict__ We1P,
                                                  const float* __restrict__ b_e1,
                                                  const float* __restrict__ W_e2,
                                                  const float* __restrict__ b_e2,
                                                  const float* __restrict__ hv,
                                                  _Float16* __restrict__ he1F16,
                                                  float* __restrict__ lg,
                                                  unsigned* __restrict__ segmaxK) {
    __shared__ _Float16 As[16 * KE1];
    __shared__ float lsum[16];
    int tile = blockIdx.x, t = threadIdx.x, ebase = tile * 16;
    if (t < 16) lsum[t] = b_e2[0];
    __syncthreads();
    // hv_new[dst] . W_e2[0:256]  (16 rows x 16 chunks of 16)
    {
        int r = t >> 4, cb = (t & 15) * 16;
        const float* hrow = hv + (size_t)dst[ebase + r] * GDIM + cb;
        float p = 0.f;
#pragma unroll
        for (int j = 0; j < 16; ++j) p += hrow[j] * W_e2[cb + j];
        atomicAdd(&lsum[r], p);
    }
    // gather A = [node_feats[src] | edge_feats | 0-pad] as f16
    for (int i = t; i < 16 * KE1; i += 256) {
        int r = i / KE1, col = i - r * KE1;
        int e = ebase + r;
        _Float16 v = (_Float16)0.f;
        if (col < F_NODE)       v = nodeF16[(size_t)src[e] * F_NODE + col];
        else if (col < KE1_TRUE) v = (_Float16)edge_feats[(size_t)e * F_EDGE + (col - F_NODE)];
        As[i] = v;
    }
    __syncthreads();
    int wave = t >> 5, lane = t & 31;
    const _Float16* arow = &As[(lane & 15) * KE1];
    v16h afrag[5];
#pragma unroll
    for (int kt = 0; kt < 5; ++kt) afrag[kt] = load_a_frag(arow + kt * 32, lane);
    const v16h* Bp = (const v16h*)We1P;
#pragma unroll
    for (int ni = 0; ni < 2; ++ni) {
        int nt = wave * 2 + ni;
        v8f acc = {0.f,0.f,0.f,0.f,0.f,0.f,0.f,0.f};
#pragma unroll
        for (int kt = 0; kt < 5; ++kt)
            acc = wmma_f16(afrag[kt], Bp[(kt * 16 + nt) * 32 + lane], acc);
        int n = nt * 16 + (lane & 15);
        float bias = b_e1[n];
        float w2 = W_e2[GDIM + n];
#pragma unroll
        for (int r = 0; r < 8; ++r) {
            int mrow = r + ((lane & 16) ? 8 : 0);
            float v = lrelu(acc[r] + bias);
            he1F16[(size_t)(ebase + mrow) * GDIM + n] = (_Float16)v;
            atomicAdd(&lsum[mrow], v * w2);
        }
    }
    __syncthreads();
    if (t < 16) {
        int e = ebase + t;
        float L = lrelu(lsum[t]);
        lg[e] = L;
        atomicMax(&segmaxK[dst[e]], fenc(L));
    }
}

// ---------------- stage 4: segment softmax numerator + denominator ----------------
__global__ void k_edge_soft(const float* __restrict__ lg, const int* __restrict__ dst,
                            const unsigned* __restrict__ segmaxK,
                            float* __restrict__ ex, float* __restrict__ segsum) {
    int e = blockIdx.x * blockDim.x + threadIdx.x;
    if (e >= N_EDGES) return;
    int d = dst[e];
    float x = __expf(lg[e] - fdec(segmaxK[d]));
    ex[e] = x;
    atomicAdd(&segsum[d], x);
}

// ---------------- stage 5+6: e = alpha*(he1 @ W_et + b), segment-sum into c ----------------
__global__ void __launch_bounds__(256) k_edge_et(const _Float16* __restrict__ he1F16,
                                                 const int* __restrict__ dst,
                                                 const _Float16* __restrict__ WetP,
                                                 const float* __restrict__ b_et,
                                                 const float* __restrict__ ex,
                                                 const float* __restrict__ segsum,
                                                 float* __restrict__ c) {
    __shared__ float alph[16];
    __shared__ int dstl[16];
    int tile = blockIdx.x, t = threadIdx.x, ebase = tile * 16;
    if (t < 16) {
        int e = ebase + t, d = dst[e];
        dstl[t] = d;
        alph[t] = ex[e] / fmaxf(segsum[d], 1e-12f);
    }
    __syncthreads();
    int wave = t >> 5, lane = t & 31;
    const _Float16* arow = he1F16 + (size_t)(ebase + (lane & 15)) * GDIM;
    v16h afrag[8];
#pragma unroll
    for (int kt = 0; kt < 8; ++kt) afrag[kt] = load_a_frag(arow + kt * 32, lane);
    const v16h* Bp = (const v16h*)WetP;
#pragma unroll
    for (int ni = 0; ni < 2; ++ni) {
        int nt = wave * 2 + ni;
        v8f acc = {0.f,0.f,0.f,0.f,0.f,0.f,0.f,0.f};
#pragma unroll
        for (int kt = 0; kt < 8; ++kt)
            acc = wmma_f16(afrag[kt], Bp[(kt * 16 + nt) * 32 + lane], acc);
        int n = nt * 16 + (lane & 15);
        float bias = b_et[n];
#pragma unroll
        for (int r = 0; r < 8; ++r) {
            int mrow = r + ((lane & 16) ? 8 : 0);
            atomicAdd(&c[(size_t)dstl[mrow] * GDIM + n], alph[mrow] * (acc[r] + bias));
        }
    }
}

// ---------------- stage 7: GRU cell ----------------
__global__ void __launch_bounds__(256) k_gru(const float* __restrict__ cbuf,
                                             const float* __restrict__ hv,
                                             const _Float16* __restrict__ WihP,
                                             const _Float16* __restrict__ WhhP,
                                             const float* __restrict__ b_ih,
                                             const float* __restrict__ b_hh,
                                             float* __restrict__ out) {
    __shared__ _Float16 Ac[16 * GDIM];
    __shared__ _Float16 Ah[16 * GDIM];
    int tile = blockIdx.x, t = threadIdx.x;
    for (int i = t; i < 16 * GDIM; i += 256) {
        int r = i >> 8, col = i & 255;
        size_t idx = (size_t)(tile * 16 + r) * GDIM + col;
        float cv = cbuf[idx];
        Ac[i] = (_Float16)(cv > 0.f ? cv : (__expf(cv) - 1.f));  // elu
        Ah[i] = (_Float16)hv[idx];
    }
    __syncthreads();
    int wave = t >> 5, lane = t & 31;
    const _Float16* arc = &Ac[(lane & 15) * GDIM];
    const _Float16* arh = &Ah[(lane & 15) * GDIM];
    v16h ac[8], ah[8];
#pragma unroll
    for (int kt = 0; kt < 8; ++kt) {
        ac[kt] = load_a_frag(arc + kt * 32, lane);
        ah[kt] = load_a_frag(arh + kt * 32, lane);
    }
    const v16h* Bi = (const v16h*)WihP;   // K=256, N=768 -> 48 ntiles
    const v16h* Bh = (const v16h*)WhhP;
#pragma unroll
    for (int ni = 0; ni < 2; ++ni) {
        int nt = wave * 2 + ni;
        v8f air = {0.f,0.f,0.f,0.f,0.f,0.f,0.f,0.f}, aiz = air, ain = air;
        v8f ahr = air, ahz = air, ahn = air;
#pragma unroll
        for (int kt = 0; kt < 8; ++kt) {
            air = wmma_f16(ac[kt], Bi[(kt * 48 + nt     ) * 32 + lane], air);
            aiz = wmma_f16(ac[kt], Bi[(kt * 48 + nt + 16) * 32 + lane], aiz);
            ain = wmma_f16(ac[kt], Bi[(kt * 48 + nt + 32) * 32 + lane], ain);
            ahr = wmma_f16(ah[kt], Bh[(kt * 48 + nt     ) * 32 + lane], ahr);
            ahz = wmma_f16(ah[kt], Bh[(kt * 48 + nt + 16) * 32 + lane], ahz);
            ahn = wmma_f16(ah[kt], Bh[(kt * 48 + nt + 32) * 32 + lane], ahn);
        }
        int n = nt * 16 + (lane & 15);
        float bir = b_ih[n], biz = b_ih[GDIM + n], bin = b_ih[2 * GDIM + n];
        float bhr = b_hh[n], bhz = b_hh[GDIM + n], bhn = b_hh[2 * GDIM + n];
#pragma unroll
        for (int r = 0; r < 8; ++r) {
            int m = tile * 16 + r + ((lane & 16) ? 8 : 0);
            float rg = sigmoidf(air[r] + bir + ahr[r] + bhr);
            float zg = sigmoidf(aiz[r] + biz + ahz[r] + bhz);
            float ng = tanhf(ain[r] + bin + rg * (ahn[r] + bhn));
            float hvv = hv[(size_t)m * GDIM + n];
            float ho = (1.f - zg) * ng + zg * hvv;
            out[(size_t)m * GDIM + n] = fmaxf(ho, 0.f);
        }
    }
}

// ---------------- host launcher ----------------
extern "C" void kernel_launch(void* const* d_in, const int* in_sizes, int n_in,
                              void* d_out, int out_size, void* d_ws, size_t ws_size,
                              hipStream_t stream) {
    const float* node_feats = (const float*)d_in[0];
    const float* edge_feats = (const float*)d_in[1];
    const int*   src        = (const int*)  d_in[2];
    const int*   dst        = (const int*)  d_in[3];
    const float* W_node     = (const float*)d_in[4];
    const float* b_node     = (const float*)d_in[5];
    const float* W_e1       = (const float*)d_in[6];
    const float* b_e1       = (const float*)d_in[7];
    const float* W_e2       = (const float*)d_in[8];
    const float* b_e2       = (const float*)d_in[9];
    const float* W_et       = (const float*)d_in[10];
    const float* b_et       = (const float*)d_in[11];
    const float* W_ih       = (const float*)d_in[12];
    const float* W_hh       = (const float*)d_in[13];
    const float* b_ih       = (const float*)d_in[14];
    const float* b_hh       = (const float*)d_in[15];
    float* out = (float*)d_out;

    // workspace bump allocator (256B aligned)
    uint8_t* p = (uint8_t*)d_ws;
    auto alloc = [&](size_t bytes) -> void* {
        void* r = (void*)p;
        p += (bytes + 255) & ~(size_t)255;
        return r;
    };
    _Float16* nodeF16 = (_Float16*)alloc((size_t)N_NODES * F_NODE * 2);
    _Float16* WnodeP  = (_Float16*)alloc((size_t)F_NODE * GDIM * 2);
    _Float16* We1P    = (_Float16*)alloc((size_t)KE1 * GDIM * 2);
    _Float16* WetP    = (_Float16*)alloc((size_t)GDIM * GDIM * 2);
    _Float16* WihP    = (_Float16*)alloc((size_t)GDIM * 3 * GDIM * 2);
    _Float16* WhhP    = (_Float16*)alloc((size_t)GDIM * 3 * GDIM * 2);
    float*    hv      = (float*)   alloc((size_t)N_NODES * GDIM * 4);
    _Float16* he1F16  = (_Float16*)alloc((size_t)N_EDGES * GDIM * 2);
    float*    lg      = (float*)   alloc((size_t)N_EDGES * 4);
    float*    ex      = (float*)   alloc((size_t)N_EDGES * 4);
    unsigned* segmaxK = (unsigned*)alloc((size_t)N_NODES * 4);
    float*    segsum  = (float*)   alloc((size_t)N_NODES * 4);
    float*    cbuf    = (float*)   alloc((size_t)N_NODES * GDIM * 4);

    // pack weights / convert node feats to f16 (tiny one-time cost per launch)
    {
        int n = N_NODES * F_NODE;
        k_cvt_f16<<<(n + 255) / 256, 256, 0, stream>>>(node_feats, nodeF16, n);
    }
    k_pack_b<<<(F_NODE * GDIM + 255) / 256, 256, 0, stream>>>(W_node, WnodeP, F_NODE, F_NODE, GDIM, 0);
    k_pack_b<<<(KE1 * GDIM + 255) / 256, 256, 0, stream>>>(W_e1, We1P, KE1, KE1_TRUE, GDIM, 0);
    k_pack_b<<<(GDIM * GDIM + 255) / 256, 256, 0, stream>>>(W_et, WetP, GDIM, GDIM, GDIM, 0);
    k_pack_b<<<(GDIM * 3 * GDIM + 255) / 256, 256, 0, stream>>>(W_ih, WihP, GDIM, GDIM, 3 * GDIM, 1);
    k_pack_b<<<(GDIM * 3 * GDIM + 255) / 256, 256, 0, stream>>>(W_hh, WhhP, GDIM, GDIM, 3 * GDIM, 1);
    k_init<<<(N_NODES * GDIM + 255) / 256, 256, 0, stream>>>(segmaxK, segsum, cbuf);

    // stage 1: node GEMM
    k_node_gemm<<<N_NODES / 16, 256, 0, stream>>>(nodeF16, WnodeP, b_node, hv);
    // stage 2+3: edge MLP + logits + segment max
    k_edge_he1<<<N_EDGES / 16, 256, 0, stream>>>(nodeF16, edge_feats, src, dst, We1P,
                                                 b_e1, W_e2, b_e2, hv, he1F16, lg, segmaxK);
    // stage 4: softmax numerators + segment sums
    k_edge_soft<<<N_EDGES / 256, 256, 0, stream>>>(lg, dst, segmaxK, ex, segsum);
    // stage 5+6: edge transform + weighted segment sum
    k_edge_et<<<N_EDGES / 16, 256, 0, stream>>>(he1F16, dst, WetP, b_et, ex, segsum, cbuf);
    // stage 7: GRU
    k_gru<<<N_NODES / 16, 256, 0, stream>>>(cbuf, hv, WihP, WhhP, b_ih, b_hh, out);
}
